// SparseMoE_10720238371033
// MI455X (gfx1250) — matrix operand order repaired and checked by reading, not compiled, not run on hardware
//
#include <hip/hip_runtime.h>

// ---------------- problem constants (from setup_inputs) ----------------
#define TT 4096   // tokens = N*P
#define DD 1024   // hidden dim
#define EE 8      // experts
#define HH 2048   // expert output dim

// ws layout:
//   [0    ..   31] int   cnt[8]
//   [32   ..   63] float sum[8]
//   [64   ..   95] float sumsq[8]
//   [96   ..   99] float invk (1/top_k, written by gate kernel)
//   [128  .. 4223] zero row (read as fp32 or bf16 for invalid token rows)
//   [4224 .. +128KB] int lists[8][TT]
//   [135424 .. +8MB]  bf16 x       (big-ws path)
//   [8524032 .. +32MB] bf16 expert_W (big-ws path)
#define WS_OFF_ZERO   128
#define WS_OFF_LISTS  4224
#define WS_OFF_XBF    135424
#define WS_OFF_WBF    8524032
#define WS_NEED       42078464ull

typedef __attribute__((ext_vector_type(16))) __bf16 v16bf;
typedef __attribute__((ext_vector_type(8)))  float  v8f;

union BF16x16 { uint4 q[2]; v16bf v; };   // 32B fragment: two b128 loads

__device__ __forceinline__ v16bf pack16(float4 q0, float4 q1, float4 q2, float4 q3) {
    v16bf v;
    v[0]  = (__bf16)q0.x; v[1]  = (__bf16)q0.y; v[2]  = (__bf16)q0.z; v[3]  = (__bf16)q0.w;
    v[4]  = (__bf16)q1.x; v[5]  = (__bf16)q1.y; v[6]  = (__bf16)q1.z; v[7]  = (__bf16)q1.w;
    v[8]  = (__bf16)q2.x; v[9]  = (__bf16)q2.y; v[10] = (__bf16)q2.z; v[11] = (__bf16)q2.w;
    v[12] = (__bf16)q3.x; v[13] = (__bf16)q3.y; v[14] = (__bf16)q3.z; v[15] = (__bf16)q3.w;
    return v;
}

// ---------------- fp32 -> bf16 streaming conversion (8 elems/thread) ----------------
__global__ __launch_bounds__(256)
void cvt_bf16_kernel(const float* __restrict__ src, __bf16* __restrict__ dst, int n8) {
    int i = blockIdx.x * 256 + threadIdx.x;
    if (i >= n8) return;
    const float4* s4 = (const float4*)(src + (size_t)i * 8);
    float4 a = s4[0], b = s4[1];
    union { __bf16 h[8]; uint4 u; } r;
    r.h[0] = (__bf16)a.x; r.h[1] = (__bf16)a.y; r.h[2] = (__bf16)a.z; r.h[3] = (__bf16)a.w;
    r.h[4] = (__bf16)b.x; r.h[5] = (__bf16)b.y; r.h[6] = (__bf16)b.z; r.h[7] = (__bf16)b.w;
    *(uint4*)(dst + (size_t)i * 8) = r.u;
}

// ---------------- kernel 1: gating + top-2 routing + variance stats ----------------
__global__ __launch_bounds__(256)
void moe_gate_kernel(const float* __restrict__ x,
                     const float* __restrict__ gW,
                     const float* __restrict__ gb,
                     const int*   __restrict__ topk,
                     int*   __restrict__ cnt,
                     float* __restrict__ sum,
                     float* __restrict__ sumsq,
                     float* __restrict__ invk_out,
                     int*   __restrict__ lists)
{
    if (blockIdx.x == 0 && threadIdx.x == 0) *invk_out = 1.0f / (float)(*topk);

    const int lane = threadIdx.x & 31;
    const int wave = threadIdx.x >> 5;
    const int t    = blockIdx.x * 8 + wave;          // one wave32 per token
    const float* xr = x + (size_t)t * DD;

    float acc[EE];
#pragma unroll
    for (int e = 0; e < EE; ++e) acc[e] = 0.0f;
    for (int k = lane; k < DD; k += 32) {            // gW (32KB) stays in L0
        float xv = xr[k];
#pragma unroll
        for (int e = 0; e < EE; ++e) acc[e] += xv * gW[e * DD + k];
    }
#pragma unroll
    for (int e = 0; e < EE; ++e)
#pragma unroll
        for (int off = 16; off > 0; off >>= 1)
            acc[e] += __shfl_xor(acc[e], off, 32);

    if (lane == 0) {
        float logit[EE], p[EE];
        float m = -3.4e38f;
#pragma unroll
        for (int e = 0; e < EE; ++e) { logit[e] = acc[e] + gb[e]; m = fmaxf(m, logit[e]); }
        float s = 0.0f;
#pragma unroll
        for (int e = 0; e < EE; ++e) { p[e] = __expf(logit[e] - m); s += p[e]; }
        float inv = 1.0f / s;
#pragma unroll
        for (int e = 0; e < EE; ++e) p[e] *= inv;

        int i1 = 0; float b1 = p[0];
#pragma unroll
        for (int e = 1; e < EE; ++e) if (p[e] > b1) { b1 = p[e]; i1 = e; }
        int i2 = (i1 == 0) ? 1 : 0; float b2 = -1.0f;
#pragma unroll
        for (int e = 0; e < EE; ++e) if (e != i1 && p[e] > b2) { b2 = p[e]; i2 = e; }

        int p1 = atomicAdd(&cnt[i1], 1); lists[i1 * TT + p1] = t;
        int p2 = atomicAdd(&cnt[i2], 1); lists[i2 * TT + p2] = t;
#pragma unroll
        for (int e = 0; e < EE; ++e) {
            atomicAdd(&sum[e],   p[e]);
            atomicAdd(&sumsq[e], p[e] * p[e]);
        }
    }
}

// ---------------- kernel 2a: gathered per-expert bf16 WMMA GEMM (pre-converted) ----
// grid = (HH/128, TT/64, EE); block = 256 = 8 waves (2 M x 4 N), wave tile 32x32.
__global__ __launch_bounds__(256)
void moe_expert_gemm_bf16(const __bf16* __restrict__ xbf,
                          const __bf16* __restrict__ wbf,
                          const float*  __restrict__ eb,
                          const int*    __restrict__ cnt,
                          const int*    __restrict__ lists,
                          const float*  __restrict__ invk_p,
                          const __bf16* __restrict__ zerorow,
                          float* __restrict__ out)
{
    const int e = blockIdx.z;
    const int nt = cnt[e];
    const int tile0 = blockIdx.y * 64;
    if (tile0 >= nt) return;                          // uniform exit keeps EXEC all-1s

    __shared__ int s_tok[64];
    const int tid = threadIdx.x;
    if (tid < 64) { int g = tile0 + tid; s_tok[tid] = (g < nt) ? lists[e * TT + g] : -1; }
    __syncthreads();

    const int lane = tid & 31, wave = tid >> 5;
    const int nw = wave >> 1, mw = wave & 1;
    const int hl = lane >> 4, lc = lane & 15;
    const int nbase = blockIdx.x * 128 + nw * 32;

    const __bf16* We = wbf + (size_t)e * HH * DD;
    const __bf16* brow0 = We + (size_t)(nbase      + lc) * DD + hl * 16;  // B frag: K=16*hl+i
    const __bf16* brow1 = We + (size_t)(nbase + 16 + lc) * DD + hl * 16;

    const int tk0 = s_tok[mw * 32 + lc];
    const int tk1 = s_tok[mw * 32 + 16 + lc];
    const __bf16* ar0 = ((tk0 < 0) ? zerorow : xbf + (size_t)tk0 * DD) + hl * 16;  // A: K=16*hl+i
    const __bf16* ar1 = ((tk1 < 0) ? zerorow : xbf + (size_t)tk1 * DD) + hl * 16;

    v8f c00 = {}, c01 = {}, c10 = {}, c11 = {};
    for (int kb = 0; kb < DD; kb += 32) {
        BF16x16 a0, a1, b0, b1;
        a0.q[0] = *(const uint4*)(ar0 + kb);      a0.q[1] = *(const uint4*)(ar0 + kb + 8);
        a1.q[0] = *(const uint4*)(ar1 + kb);      a1.q[1] = *(const uint4*)(ar1 + kb + 8);
        b0.q[0] = *(const uint4*)(brow0 + kb);    b0.q[1] = *(const uint4*)(brow0 + kb + 8);
        b1.q[0] = *(const uint4*)(brow1 + kb);    b1.q[1] = *(const uint4*)(brow1 + kb + 8);
        c00 = __builtin_amdgcn_wmma_f32_16x16x32_bf16(false, a0.v, false, b0.v, (short)0, c00, false, false);
        c01 = __builtin_amdgcn_wmma_f32_16x16x32_bf16(false, a0.v, false, b1.v, (short)0, c01, false, false);
        c10 = __builtin_amdgcn_wmma_f32_16x16x32_bf16(false, a1.v, false, b0.v, (short)0, c10, false, false);
        c11 = __builtin_amdgcn_wmma_f32_16x16x32_bf16(false, a1.v, false, b1.v, (short)0, c11, false, false);
    }

    const float invk = *invk_p;
    const int h0 = nbase + lc, h1 = nbase + 16 + lc;
    const float bi0 = invk * eb[e * HH + h0];
    const float bi1 = invk * eb[e * HH + h1];
#pragma unroll
    for (int r = 0; r < 8; ++r) {
        const int mloc = r + 8 * hl;                  // C/D: VGPR r -> M = r + 8*half
        int t0 = s_tok[mw * 32 + mloc];
        if (t0 >= 0) {
            float* o = out + (size_t)t0 * HH;
            atomicAdd(o + h0, invk * c00[r] + bi0);
            atomicAdd(o + h1, invk * c01[r] + bi1);
        }
        int t1 = s_tok[mw * 32 + 16 + mloc];
        if (t1 >= 0) {
            float* o = out + (size_t)t1 * HH;
            atomicAdd(o + h0, invk * c10[r] + bi0);
            atomicAdd(o + h1, invk * c11[r] + bi1);
        }
    }
}

// ---------------- kernel 2b: fallback (fp32 in, in-loop cvt), same tiling ----------
__global__ __launch_bounds__(256)
void moe_expert_gemm_f32(const float* __restrict__ x,
                         const float* __restrict__ eW,
                         const float* __restrict__ eb,
                         const int*   __restrict__ cnt,
                         const int*   __restrict__ lists,
                         const float* __restrict__ invk_p,
                         const float* __restrict__ zerorow,
                         float* __restrict__ out)
{
    const int e = blockIdx.z;
    const int nt = cnt[e];
    const int tile0 = blockIdx.y * 64;
    if (tile0 >= nt) return;

    __shared__ int s_tok[64];
    const int tid = threadIdx.x;
    if (tid < 64) { int g = tile0 + tid; s_tok[tid] = (g < nt) ? lists[e * TT + g] : -1; }
    __syncthreads();

    const int lane = tid & 31, wave = tid >> 5;
    const int nw = wave >> 1, mw = wave & 1;
    const int hl = lane >> 4, lc = lane & 15;
    const int nbase = blockIdx.x * 128 + nw * 32;

    const float* We = eW + (size_t)e * HH * DD;
    const float* brow0 = We + (size_t)(nbase      + lc) * DD + hl * 16;
    const float* brow1 = We + (size_t)(nbase + 16 + lc) * DD + hl * 16;
    const int tk0 = s_tok[mw * 32 + lc];
    const int tk1 = s_tok[mw * 32 + 16 + lc];
    const float* ar0 = ((tk0 < 0) ? zerorow : x + (size_t)tk0 * DD) + hl * 16;
    const float* ar1 = ((tk1 < 0) ? zerorow : x + (size_t)tk1 * DD) + hl * 16;

    v8f c00 = {}, c01 = {}, c10 = {}, c11 = {};
    for (int kb = 0; kb < DD; kb += 32) {
        const float4* b04 = (const float4*)(brow0 + kb);
        const float4* b14 = (const float4*)(brow1 + kb);
        const float4* a04 = (const float4*)(ar0 + kb);
        const float4* a14 = (const float4*)(ar1 + kb);
        v16bf bf0 = pack16(b04[0], b04[1], b04[2], b04[3]);
        v16bf bf1 = pack16(b14[0], b14[1], b14[2], b14[3]);
        v16bf af0 = pack16(a04[0], a04[1], a04[2], a04[3]);
        v16bf af1 = pack16(a14[0], a14[1], a14[2], a14[3]);
        c00 = __builtin_amdgcn_wmma_f32_16x16x32_bf16(false, af0, false, bf0, (short)0, c00, false, false);
        c01 = __builtin_amdgcn_wmma_f32_16x16x32_bf16(false, af0, false, bf1, (short)0, c01, false, false);
        c10 = __builtin_amdgcn_wmma_f32_16x16x32_bf16(false, af1, false, bf0, (short)0, c10, false, false);
        c11 = __builtin_amdgcn_wmma_f32_16x16x32_bf16(false, af1, false, bf1, (short)0, c11, false, false);
    }

    const float invk = *invk_p;
    const int h0 = nbase + lc, h1 = nbase + 16 + lc;
    const float bi0 = invk * eb[e * HH + h0];
    const float bi1 = invk * eb[e * HH + h1];
#pragma unroll
    for (int r = 0; r < 8; ++r) {
        const int mloc = r + 8 * hl;
        int t0 = s_tok[mw * 32 + mloc];
        if (t0 >= 0) {
            float* o = out + (size_t)t0 * HH;
            atomicAdd(o + h0, invk * c00[r] + bi0);
            atomicAdd(o + h1, invk * c01[r] + bi1);
        }
        int t1 = s_tok[mw * 32 + 16 + mloc];
        if (t1 >= 0) {
            float* o = out + (size_t)t1 * HH;
            atomicAdd(o + h0, invk * c10[r] + bi0);
            atomicAdd(o + h1, invk * c11[r] + bi1);
        }
    }
}

// ---------------- kernel 3: unbiased variance finalize ----------------
__global__ void moe_var_kernel(const float* __restrict__ sum,
                               const float* __restrict__ sumsq,
                               float* __restrict__ var_out)
{
    int e = threadIdx.x;
    if (e < EE) {
        float s = sum[e], ss = sumsq[e], Tn = (float)TT;
        var_out[e] = (ss - s * s / Tn) / (Tn - 1.0f);
    }
}

extern "C" void kernel_launch(void* const* d_in, const int* in_sizes, int n_in,
                              void* d_out, int out_size, void* d_ws, size_t ws_size,
                              hipStream_t stream) {
    (void)in_sizes; (void)n_in;
    const float* x  = (const float*)d_in[0];
    const float* gW = (const float*)d_in[1];
    const float* gb = (const float*)d_in[2];
    const float* eW = (const float*)d_in[3];
    const float* eb = (const float*)d_in[4];
    const int*   tk = (const int*)d_in[5];
    float* out = (float*)d_out;

    char*  ws    = (char*)d_ws;
    int*   cnt   = (int*)ws;
    float* sum   = (float*)(ws + 32);
    float* sumsq = (float*)(ws + 64);
    float* invk  = (float*)(ws + 96);
    void*  zrow  = (void*)(ws + WS_OFF_ZERO);
    int*   lists = (int*)(ws + WS_OFF_LISTS);
    __bf16* xbf  = (__bf16*)(ws + WS_OFF_XBF);
    __bf16* wbf  = (__bf16*)(ws + WS_OFF_WBF);

    // zero header + zero-row each launch (graph-replay safe, capture-legal)
    hipMemsetAsync(d_ws, 0, WS_OFF_LISTS, stream);
    hipMemsetAsync(d_out, 0, (size_t)out_size * sizeof(float), stream);

    const bool big = (ws_size >= WS_NEED);
    if (big) {
        int n8x = (TT * DD) / 8, n8w = (EE * HH * DD) / 8;
        cvt_bf16_kernel<<<(n8x + 255) / 256, 256, 0, stream>>>(x,  xbf, n8x);
        cvt_bf16_kernel<<<(n8w + 255) / 256, 256, 0, stream>>>(eW, wbf, n8w);
    }

    moe_gate_kernel<<<TT / 8, 256, 0, stream>>>(x, gW, gb, tk, cnt, sum, sumsq, invk, lists);

    dim3 grid(HH / 128, TT / 64, EE);
    if (big) {
        moe_expert_gemm_bf16<<<grid, 256, 0, stream>>>(xbf, wbf, eb, cnt, lists, invk,
                                                       (const __bf16*)zrow, out);
    } else {
        moe_expert_gemm_f32<<<grid, 256, 0, stream>>>(x, eW, eb, cnt, lists, invk,
                                                      (const float*)zrow, out);
    }

    moe_var_kernel<<<1, 32, 0, stream>>>(sum, sumsq, out + (size_t)TT * HH);
}
